// SimpleBlock2dFactorizedParallel_87351044866503
// MI455X (gfx1250) — matrix operand order repaired and checked by reading, not compiled, not run on hardware
//
#include <hip/hip_runtime.h>
#include <hip/hip_bf16.h>

// ---------------------------------------------------------------------------
// SimpleBlock2dFactorizedParallel (factorized FNO) on CDNA5 (gfx1250).
// All GEMMs run on v_wmma_f32_16x16x32_f16 (wave32, f32 accumulate).
// Truncated rfft/irfft (16 modes) are dense DFT matmuls. Shared activation
// tiles are staged to LDS via GLOBAL_LOAD_ASYNC_TO_LDS_B128 when available.
// Forecast head is algebraically folded: (f@W1+b1)@W2+b2 = f@(W1@W2)+const.
// ---------------------------------------------------------------------------

typedef _Float16 half_t;
typedef __attribute__((ext_vector_type(16))) half_t v16h;
typedef __attribute__((ext_vector_type(8)))  float  v8f;

#define WMMA_F16(a, b, c) \
  __builtin_amdgcn_wmma_f32_16x16x32_f16(false, (a), false, (b), (short)0, (c), false, false)

#define B_SZ 16
#define M_SZ 128
#define N_SZ 128
#define D_SZ 64
#define H_SZ 128
#define MODES 16
#define NPTS (B_SZ * M_SZ * N_SZ)          // 262144 spatial points
#define ROWS2D (B_SZ * M_SZ)               // 2048 (b, other-axis) rows
#define PLANE (M_SZ * N_SZ * D_SZ)         // 1048576 floats per batch image
#define FF_R 4                             // row tiles per wave in FF kernel

#ifndef __has_builtin
#define __has_builtin(x) 0
#endif
#if __has_builtin(__builtin_amdgcn_global_load_async_to_lds_b128)
#define ASYNC_LDS 1
#else
#define ASYNC_LDS 0
#endif

#if ASYNC_LDS
typedef int v4i __attribute__((vector_size(16)));
typedef __attribute__((address_space(1))) v4i* as1_v4i;
typedef __attribute__((address_space(3))) v4i* as3_v4i;
#endif

// Coalesced 16B global->LDS copy (async when the CDNA5 builtin exists).
static __device__ __forceinline__ void copy16B(const float* g, float* l) {
#if ASYNC_LDS
  __builtin_amdgcn_global_load_async_to_lds_b128((as1_v4i)g, (as3_v4i)l, 0, 0);
#else
  *(float4*)l = *(const float4*)g;
#endif
}

static __device__ __forceinline__ void async_join() {
#if ASYNC_LDS
#if __has_builtin(__builtin_amdgcn_s_wait_asynccnt)
  __builtin_amdgcn_s_wait_asynccnt(0);
#else
  asm volatile("s_wait_asynccnt 0x0" ::: "memory");
#endif
#endif
}

static __device__ __forceinline__ int lane_id() { return (int)(threadIdx.x & 31); }

static __device__ __forceinline__ v8f zero8() {
  v8f c;
#pragma unroll
  for (int i = 0; i < 8; ++i) c[i] = 0.0f;
  return c;
}

static __device__ __forceinline__ v16h neg16(v16h a) {
  v16h r;
#pragma unroll
  for (int e = 0; e < 16; ++e) r[e] = (half_t)(-(float)a[e]);
  return r;
}

// --- A fragment (16x32 tile), element (r,k) = p[r*rs + k] -------------------
// lanes 0-15: row M=lane, K in {0..15}; lanes 16-31: row M=lane-16,
// K in {16..31}; element e -> K = kb + e.
static __device__ __forceinline__ v16h frag_a_f32(const float* p, int rs) {
  int lane = lane_id();
  const float* row = p + (size_t)(lane & 15) * (size_t)rs;
  int kb = (lane & 16) ? 16 : 0;
  v16h a;
#pragma unroll
  for (int e = 0; e < 16; ++e) {
    int k = kb + e;
    a[e] = (half_t)row[k];
  }
  return a;
}

static __device__ __forceinline__ v16h frag_a_f32_pad(const float* p, int rs, int kmax) {
  int lane = lane_id();
  const float* row = p + (size_t)(lane & 15) * (size_t)rs;
  int kb = (lane & 16) ? 16 : 0;
  v16h a;
#pragma unroll
  for (int e = 0; e < 16; ++e) {
    int k = kb + e;
    a[e] = (k < kmax) ? (half_t)row[k] : (half_t)0.0f;
  }
  return a;
}

static __device__ __forceinline__ v16h frag_a_f16(const half_t* p, int rs) {
  int lane = lane_id();
  const half_t* row = p + (size_t)(lane & 15) * (size_t)rs;
  int kb = (lane & 16) ? 16 : 0;
  v16h a;
#pragma unroll
  for (int e = 0; e < 16; ++e) {
    int k = kb + e;
    a[e] = row[k];
  }
  return a;
}

// --- B fragment (32x16 tile), element (k,n) = p[k*rs + n] -------------------
// lanes 0-15 hold K=0..15 (col N=lane), lanes 16-31 hold K=16..31.
static __device__ __forceinline__ v16h frag_b_f16(const half_t* p, int rs) {
  int lane = lane_id();
  int n = lane & 15;
  int kb = (lane & 16) ? 16 : 0;
  v16h b;
#pragma unroll
  for (int e = 0; e < 16; ++e) b[e] = p[(size_t)(kb + e) * (size_t)rs + n];
  return b;
}

static __device__ __forceinline__ v16h frag_b_f32(const float* p, int rs) {
  int lane = lane_id();
  int n = lane & 15;
  int kb = (lane & 16) ? 16 : 0;
  v16h b;
#pragma unroll
  for (int e = 0; e < 16; ++e) b[e] = (half_t)p[(size_t)(kb + e) * (size_t)rs + n];
  return b;
}

// B fragment: K=0..15 from plo, K=16..31 from phi (iDFT stacked Re;Im)
static __device__ __forceinline__ v16h frag_b_dual_f32(const float* plo, const float* phi, int rs) {
  int lane = lane_id();
  int n = lane & 15;
  const float* p = (lane & 16) ? phi : plo;
  v16h b;
#pragma unroll
  for (int e = 0; e < 16; ++e) b[e] = (half_t)p[(size_t)e * (size_t)rs + n];
  return b;
}

// --- C/D fragment: d[r] <-> (M = r + 8*(lane>=16), N = lane&15) -------------
static __device__ __forceinline__ v8f frag_c_bias(const float* bias) {
  float v = bias[lane_id() & 15];
  v8f c;
#pragma unroll
  for (int r = 0; r < 8; ++r) c[r] = v;
  return c;
}

static __device__ __forceinline__ void store_d_f32(float* p, int rs, v8f c) {
  int lane = lane_id();
  int n = lane & 15;
  int mb = (lane & 16) ? 8 : 0;
#pragma unroll
  for (int r = 0; r < 8; ++r) p[(size_t)(mb + r) * (size_t)rs + n] = c[r];
}

static __device__ __forceinline__ void store_d_relu_f16(half_t* p, int rs, v8f c) {
  int lane = lane_id();
  int n = lane & 15;
  int mb = (lane & 16) ? 8 : 0;
#pragma unroll
  for (int r = 0; r < 8; ++r) {
    float v = c[r];
    p[(size_t)(mb + r) * (size_t)rs + n] = (half_t)(v > 0.0f ? v : 0.0f);
  }
}

static __device__ __forceinline__ void store_d_f16(half_t* p, int rs, v8f c) {
  int lane = lane_id();
  int n = lane & 15;
  int mb = (lane & 16) ? 8 : 0;
#pragma unroll
  for (int r = 0; r < 8; ++r) p[(size_t)(mb + r) * (size_t)rs + n] = (half_t)c[r];
}

// ===========================================================================
// Prep kernels
// ===========================================================================
__global__ void zero_f32_kernel(float* p, int n) {
  int i = blockIdx.x * blockDim.x + threadIdx.x;
  if (i < n) p[i] = 0.0f;
}

__global__ void cvt_f32_to_f16_kernel(const float* __restrict__ src, half_t* __restrict__ dst, int n) {
  int i = blockIdx.x * blockDim.x + threadIdx.x;
  if (i < n) dst[i] = (half_t)src[i];
}

// in_w [12,64] -> padded f16 [32,64]
__global__ void cvt_inw_kernel(const float* __restrict__ src, half_t* __restrict__ dst) {
  int i = blockIdx.x * blockDim.x + threadIdx.x;
  if (i < 32 * 64) {
    int r = i >> 6, c = i & 63;
    dst[i] = (r < 12) ? (half_t)src[r * 64 + c] : (half_t)0.0f;
  }
}

// fw [L=4][C=64][O=64][Y=16] f32 -> [L][Y][C][O] f16
__global__ void cvt_modes_kernel(const float* __restrict__ src, half_t* __restrict__ dst) {
  int i = blockIdx.x * blockDim.x + threadIdx.x;
  if (i < 4 * 16 * 64 * 64) {
    int l = i >> 16;
    int y = (i >> 12) & 15;
    int c = (i >> 6) & 63;
    int o = i & 63;
    dst[i] = (half_t)src[(((size_t)l * 64 + c) * 64 + o) * 16 + y];
  }
}

// Forecast head fold: headv = ow1 @ ow2 (64), headc = ob1.ow2 + ob2 (1)
__global__ void fold_head_kernel(const float* __restrict__ ow1, const float* __restrict__ ob1,
                                 const float* __restrict__ ow2, const float* __restrict__ ob2,
                                 float* __restrict__ headv, float* __restrict__ headc) {
  int c = blockIdx.x * blockDim.x + threadIdx.x;
  if (c < 64) {
    float s = 0.0f;
    for (int j = 0; j < 128; ++j) s += ow1[c * 128 + j] * ow2[j];
    headv[c] = s;
  }
  if (c == 0) {
    float s = ob2[0];
    for (int j = 0; j < 128; ++j) s += ob1[j] * ow2[j];
    headc[0] = s;
  }
}

// Forward DFT tables Cf/Sf [16][128] and inverse table Ainv [128][32]
__global__ void dft_tables_kernel(half_t* Cf, half_t* Sf, half_t* Ainv) {
  int i = blockIdx.x * blockDim.x + threadIdx.x;
  const float inv = 0.08838834764831845f;      // 1/sqrt(128)
  const float twopi = 6.283185307179586f;
  if (i < MODES * N_SZ) {
    int y = i >> 7, n = i & 127;
    float th = twopi * (float)(y * n) / 128.0f;
    Cf[i] = (half_t)(cosf(th) * inv);
    Sf[i] = (half_t)(-sinf(th) * inv);
  }
  if (i < N_SZ * 32) {
    int n = i >> 5, k = i & 31;
    int y = k & 15;
    float w = (y == 0) ? 1.0f : 2.0f;
    float th = twopi * (float)(y * n) / 128.0f;
    float v = (k < 16) ? (w * cosf(th) * inv) : (-w * sinf(th) * inv);
    Ainv[i] = (half_t)v;
  }
}

// ===========================================================================
// Input projection: X[p,0:64] = x_in[p,0:12] @ in_w + in_b
// grid = NPTS/16 x 128 threads (4 waves = 4 column tiles, shared A tile)
// ===========================================================================
__global__ __launch_bounds__(128) void input_proj_kernel(
    const float* __restrict__ xin, const half_t* __restrict__ inw,
    const float* __restrict__ inb, float* __restrict__ X) {
  __shared__ float As[16 * 12];
  int rowbase = blockIdx.x * 16;
  int tid = threadIdx.x;
  if (tid < 96) {                         // 192 floats, 2 each
    As[tid] = xin[(size_t)rowbase * 12 + tid];
    As[tid + 96] = xin[(size_t)rowbase * 12 + tid + 96];
  }
  __syncthreads();
  int ct = tid >> 5;
  v16h a = frag_a_f32_pad(As, 12, 12);
  v16h b = frag_b_f16(inw + ct * 16, 64);
  v8f c = frag_c_bias(inb + ct * 16);
  c = WMMA_F16(a, b, c);
  store_d_f32(X + (size_t)rowbase * 64 + ct * 16, 64, c);
}

// ===========================================================================
// Forward DFT (truncated rfft): F[idx, y, c] = sum_t D[y,t] * X[idx, t, c]
// Activation block [128 x 64] staged once to LDS (async), shared by 8 waves.
// grid = 2048 x 256 threads; waves 0-3: Re (4 col tiles), waves 4-7: Im.
// ===========================================================================
__global__ __launch_bounds__(256) void dft_kernel(
    const float* __restrict__ X, float* __restrict__ Fre, float* __restrict__ Fim,
    const half_t* __restrict__ Cf, const half_t* __restrict__ Sf,
    int obase_stride, int t_stride) {
  __shared__ float Xs[N_SZ * D_SZ];      // 32 KB
  int idx = blockIdx.x;
  int b = idx >> 7, o = idx & 127;
  size_t base = (size_t)b * PLANE + (size_t)o * (size_t)obase_stride;

  // stage 128 rows of 64 floats (2048 float4 / 256 threads = 8 each)
  for (int i = threadIdx.x; i < 2048; i += 256) {
    int r = i >> 4;
    int c4 = (i & 15) << 2;
    copy16B(X + base + (size_t)r * (size_t)t_stride + c4, &Xs[r * 64 + c4]);
  }
  async_join();
  __syncthreads();

  int wave = (int)(threadIdx.x >> 5);
  int ct = wave & 3;
  const half_t* Dm = (wave < 4) ? Cf : Sf;
  float* Out = (wave < 4) ? Fre : Fim;
  v8f c = zero8();
#pragma unroll
  for (int kk = 0; kk < 4; ++kk) {
    v16h a = frag_a_f16(Dm + kk * 32, 128);
    v16h bb = frag_b_f32(&Xs[kk * 32 * 64 + ct * 16], 64);
    c = WMMA_F16(a, bb, c);
  }
  store_d_f32(Out + (size_t)idx * (MODES * D_SZ) + ct * 16, D_SZ, c);
}

// ===========================================================================
// Mode mixing: Pre = Fre*Wre - Fim*Wim ; Pim = Fre*Wim + Fim*Wre
// F tiles [16 x 64] staged to LDS (async), shared by 4 waves.
// grid = (128 row tiles, 16 modes) x 128 threads (4 col tiles)
// ===========================================================================
__global__ __launch_bounds__(128) void mode_mix_kernel(
    const float* __restrict__ Fre, const float* __restrict__ Fim,
    float* __restrict__ Pre, float* __restrict__ Pim,
    const half_t* __restrict__ Wre, const half_t* __restrict__ Wim) {
  __shared__ float Are[16 * 64];
  __shared__ float Aim[16 * 64];
  int rt = blockIdx.x;
  int y = blockIdx.y;
  size_t frow = (size_t)rt * 16 * (MODES * D_SZ) + (size_t)y * D_SZ;

  for (int i = threadIdx.x; i < 256; i += 128) {   // 256 float4 per matrix
    int r = i >> 4;
    int c4 = (i & 15) << 2;
    copy16B(Fre + frow + (size_t)r * (MODES * D_SZ) + c4, &Are[r * 64 + c4]);
    copy16B(Fim + frow + (size_t)r * (MODES * D_SZ) + c4, &Aim[r * 64 + c4]);
  }
  async_join();
  __syncthreads();

  int ot = (int)(threadIdx.x >> 5);
  const half_t* wre = Wre + (size_t)y * (D_SZ * D_SZ) + ot * 16;
  const half_t* wim = Wim + (size_t)y * (D_SZ * D_SZ) + ot * 16;

  v16h are0 = frag_a_f32(Are, 64);
  v16h are1 = frag_a_f32(Are + 32, 64);
  v16h aim0 = frag_a_f32(Aim, 64);
  v16h aim1 = frag_a_f32(Aim + 32, 64);
  v16h nim0 = neg16(aim0), nim1 = neg16(aim1);

  v16h bre0 = frag_b_f16(wre, D_SZ);
  v16h bre1 = frag_b_f16(wre + 32 * D_SZ, D_SZ);
  v16h bim0 = frag_b_f16(wim, D_SZ);
  v16h bim1 = frag_b_f16(wim + 32 * D_SZ, D_SZ);

  v8f cre = zero8();
  cre = WMMA_F16(are0, bre0, cre);
  cre = WMMA_F16(are1, bre1, cre);
  cre = WMMA_F16(nim0, bim0, cre);
  cre = WMMA_F16(nim1, bim1, cre);

  v8f cim = zero8();
  cim = WMMA_F16(are0, bim0, cim);
  cim = WMMA_F16(are1, bim1, cim);
  cim = WMMA_F16(aim0, bre0, cim);
  cim = WMMA_F16(aim1, bre1, cim);

  size_t orow = (size_t)rt * 16 * (MODES * D_SZ) + (size_t)y * D_SZ + ot * 16;
  store_d_f32(Pre + orow, MODES * D_SZ, cre);
  store_d_f32(Pim + orow, MODES * D_SZ, cim);
}

// ===========================================================================
// Inverse DFT: xs[t, c] = sum_y Ainv[t, y]*Pre + Ainv[t, 16+y]*Pim
// Pre/Pim tiles [16 x 64] staged to LDS (async), shared by 8 waves.
// grid = 2048 x 256 threads (8 spatial row tiles)
// ===========================================================================
__global__ __launch_bounds__(256) void idft_kernel(
    const float* __restrict__ Pre, const float* __restrict__ Pim,
    float* __restrict__ XS, const half_t* __restrict__ Ainv,
    int obase_stride, int t_stride, int accumulate) {
  __shared__ float Ps[2][MODES * D_SZ];  // 8 KB
  int idx = blockIdx.x;
  int b = idx >> 7, o = idx & 127;
  size_t base = (size_t)b * PLANE + (size_t)o * (size_t)obase_stride;

  for (int i = threadIdx.x; i < 256; i += 256) {   // 256 float4 per matrix
    copy16B(Pre + (size_t)idx * (MODES * D_SZ) + i * 4, &Ps[0][i * 4]);
    copy16B(Pim + (size_t)idx * (MODES * D_SZ) + i * 4, &Ps[1][i * 4]);
  }
  async_join();
  __syncthreads();

  int w = (int)(threadIdx.x >> 5);
  v16h a = frag_a_f16(Ainv + (size_t)w * 16 * 32, 32);
  int lane = lane_id();
  int n = lane & 15;
  int mb = (lane & 16) ? 8 : 0;
#pragma unroll
  for (int ct = 0; ct < 4; ++ct) {
    v16h bb = frag_b_dual_f32(&Ps[0][ct * 16], &Ps[1][ct * 16], D_SZ);
    v8f c = zero8();
    c = WMMA_F16(a, bb, c);
#pragma unroll
    for (int r = 0; r < 8; ++r) {
      size_t ix = base + (size_t)(w * 16 + mb + r) * (size_t)t_stride + (size_t)(ct * 16 + n);
      if (accumulate) XS[ix] += c[r]; else XS[ix] = c[r];
    }
  }
}

// ===========================================================================
// Fused per-layer tail (4 row tiles per wave):
//   b = relu(xs@w1b+b1b)@w2b+b2b ;  X -= b
//   f = relu(xs@w1f+b1f)@w2f+b2f ;  out += f@headv + headc
// Weight B-fragments amortized over 4 tiles; XS async-staged to LDS.
// ===========================================================================
__global__ __launch_bounds__(32) void ff_fused_kernel(
    const float* __restrict__ XS, float* __restrict__ X, float* __restrict__ out,
    const half_t* __restrict__ w1b, const float* __restrict__ b1b,
    const half_t* __restrict__ w2b, const float* __restrict__ b2b,
    const half_t* __restrict__ w1f, const float* __restrict__ b1f,
    const half_t* __restrict__ w2f, const float* __restrict__ b2f,
    const float* __restrict__ headv, const float* __restrict__ headc) {
  int rowbase = blockIdx.x * (16 * FF_R);
  int lane = lane_id();
  int n = lane & 15;
  int mb = (lane & 16) ? 8 : 0;

  __shared__ float  XSs[FF_R * 16 * D_SZ];   // 16 KB input tiles
  __shared__ half_t Hs[FF_R][16 * H_SZ];     // 16 KB hidden (f16)
  __shared__ half_t Fs[FF_R][16 * D_SZ];     // 8 KB f branch (f16)

  // stage 64 rows x 64 floats (contiguous 16 KB)
  for (int i = lane; i < 1024; i += 32) {
    copy16B(XS + (size_t)rowbase * D_SZ + i * 4, &XSs[i * 4]);
  }
  async_join();
  __syncthreads();

  v16h xa0[FF_R], xa1[FF_R];
#pragma unroll
  for (int t = 0; t < FF_R; ++t) {
    xa0[t] = frag_a_f32(&XSs[t * 16 * D_SZ], D_SZ);
    xa1[t] = frag_a_f32(&XSs[t * 16 * D_SZ + 32], D_SZ);
  }

#pragma unroll
  for (int br = 0; br < 2; ++br) {
    const half_t* w1 = br ? w1f : w1b;
    const float*  b1 = br ? b1f : b1b;
    const half_t* w2 = br ? w2f : w2b;
    const float*  b2 = br ? b2f : b2b;

    // GEMM1: [16x64]x[64x128] + bias, relu -> Hs  (B frags shared by 4 tiles)
#pragma unroll
    for (int ct = 0; ct < 8; ++ct) {
      v16h bf0 = frag_b_f16(w1 + ct * 16, H_SZ);
      v16h bf1 = frag_b_f16(w1 + 32 * H_SZ + ct * 16, H_SZ);
#pragma unroll
      for (int t = 0; t < FF_R; ++t) {
        v8f c = frag_c_bias(b1 + ct * 16);
        c = WMMA_F16(xa0[t], bf0, c);
        c = WMMA_F16(xa1[t], bf1, c);
        store_d_relu_f16(&Hs[t][ct * 16], H_SZ, c);
      }
    }
    __syncthreads();

    // GEMM2: [16x128]x[128x64] + bias  (B frags shared by 4 tiles)
#pragma unroll
    for (int ct = 0; ct < 4; ++ct) {
      v16h bw0 = frag_b_f16(w2 + ct * 16, D_SZ);
      v16h bw1 = frag_b_f16(w2 + 32 * D_SZ + ct * 16, D_SZ);
      v16h bw2 = frag_b_f16(w2 + 64 * D_SZ + ct * 16, D_SZ);
      v16h bw3 = frag_b_f16(w2 + 96 * D_SZ + ct * 16, D_SZ);
#pragma unroll
      for (int t = 0; t < FF_R; ++t) {
        v16h h0 = frag_a_f16(&Hs[t][0], H_SZ);
        v16h h1 = frag_a_f16(&Hs[t][32], H_SZ);
        v16h h2 = frag_a_f16(&Hs[t][64], H_SZ);
        v16h h3 = frag_a_f16(&Hs[t][96], H_SZ);
        v8f c = frag_c_bias(b2 + ct * 16);
        c = WMMA_F16(h0, bw0, c);
        c = WMMA_F16(h1, bw1, c);
        c = WMMA_F16(h2, bw2, c);
        c = WMMA_F16(h3, bw3, c);
        if (br == 0) {
          // backcast: x = x - b
#pragma unroll
          for (int r = 0; r < 8; ++r) {
            size_t ix = (size_t)(rowbase + 16 * t + mb + r) * D_SZ + (size_t)(ct * 16 + n);
            X[ix] = X[ix] - c[r];
          }
        } else {
          store_d_f16(&Fs[t][ct * 16], D_SZ, c);
        }
      }
    }
    __syncthreads();
  }

  // folded forecast head: out += f . headv + headc
  if (lane < 16) {
    float hc = headc[0];
#pragma unroll
    for (int t = 0; t < FF_R; ++t) {
      float s = hc;
#pragma unroll 8
      for (int c = 0; c < D_SZ; ++c) s += (float)Fs[t][lane * D_SZ + c] * headv[c];
      out[rowbase + 16 * t + lane] += s;
    }
  }
}

// ===========================================================================
// Host-side orchestration
// ===========================================================================
extern "C" void kernel_launch(void* const* d_in, const int* in_sizes, int n_in,
                              void* d_out, int out_size, void* d_ws, size_t ws_size,
                              hipStream_t stream) {
  const float* x_in   = (const float*)d_in[0];
  const float* in_w   = (const float*)d_in[1];
  const float* in_b   = (const float*)d_in[2];
  const float* fwy_re = (const float*)d_in[3];
  const float* fwy_im = (const float*)d_in[4];
  const float* fwx_re = (const float*)d_in[5];
  const float* fwx_im = (const float*)d_in[6];
  const float* bc_w1  = (const float*)d_in[7];
  const float* bc_b1  = (const float*)d_in[8];
  const float* bc_w2  = (const float*)d_in[9];
  const float* bc_b2  = (const float*)d_in[10];
  const float* fc_w1  = (const float*)d_in[11];
  const float* fc_b1  = (const float*)d_in[12];
  const float* fc_w2  = (const float*)d_in[13];
  const float* fc_b2  = (const float*)d_in[14];
  const float* out_w1 = (const float*)d_in[15];
  const float* out_b1 = (const float*)d_in[16];
  const float* out_w2 = (const float*)d_in[17];
  const float* out_b2 = (const float*)d_in[18];
  float* out = (float*)d_out;

  char* ws = (char*)d_ws;
  size_t off = 0;
  auto alloc = [&](size_t bytes) -> char* {
    char* p = ws + off;
    off += (bytes + 255) & ~(size_t)255;
    return p;
  };

  float* X   = (float*)alloc((size_t)NPTS * D_SZ * 4);
  float* XS  = (float*)alloc((size_t)NPTS * D_SZ * 4);
  float* Fre = (float*)alloc((size_t)ROWS2D * MODES * D_SZ * 4);
  float* Fim = (float*)alloc((size_t)ROWS2D * MODES * D_SZ * 4);
  float* Pre = (float*)alloc((size_t)ROWS2D * MODES * D_SZ * 4);
  float* Pim = (float*)alloc((size_t)ROWS2D * MODES * D_SZ * 4);
  half_t* inw_h  = (half_t*)alloc(32 * 64 * 2);
  half_t* Wy_re  = (half_t*)alloc((size_t)4 * 16 * 64 * 64 * 2);
  half_t* Wy_im  = (half_t*)alloc((size_t)4 * 16 * 64 * 64 * 2);
  half_t* Wx_re  = (half_t*)alloc((size_t)4 * 16 * 64 * 64 * 2);
  half_t* Wx_im  = (half_t*)alloc((size_t)4 * 16 * 64 * 64 * 2);
  half_t* bcw1_h = (half_t*)alloc((size_t)4 * 64 * 128 * 2);
  half_t* bcw2_h = (half_t*)alloc((size_t)4 * 128 * 64 * 2);
  half_t* fcw1_h = (half_t*)alloc((size_t)4 * 64 * 128 * 2);
  half_t* fcw2_h = (half_t*)alloc((size_t)4 * 128 * 64 * 2);
  half_t* Cf     = (half_t*)alloc((size_t)MODES * N_SZ * 2);
  half_t* Sf     = (half_t*)alloc((size_t)MODES * N_SZ * 2);
  half_t* Ainv   = (half_t*)alloc((size_t)N_SZ * 32 * 2);
  float* headv   = (float*)alloc(64 * 4);
  float* headc   = (float*)alloc(4);
  (void)ws_size; (void)n_in; (void)in_sizes; (void)out_size;

  // --- prep ---
  zero_f32_kernel<<<(NPTS + 255) / 256, 256, 0, stream>>>(out, NPTS);
  cvt_inw_kernel<<<8, 256, 0, stream>>>(in_w, inw_h);
  cvt_modes_kernel<<<(262144 + 255) / 256, 256, 0, stream>>>(fwy_re, Wy_re);
  cvt_modes_kernel<<<(262144 + 255) / 256, 256, 0, stream>>>(fwy_im, Wy_im);
  cvt_modes_kernel<<<(262144 + 255) / 256, 256, 0, stream>>>(fwx_re, Wx_re);
  cvt_modes_kernel<<<(262144 + 255) / 256, 256, 0, stream>>>(fwx_im, Wx_im);
  cvt_f32_to_f16_kernel<<<(32768 + 255) / 256, 256, 0, stream>>>(bc_w1, bcw1_h, 32768);
  cvt_f32_to_f16_kernel<<<(32768 + 255) / 256, 256, 0, stream>>>(bc_w2, bcw2_h, 32768);
  cvt_f32_to_f16_kernel<<<(32768 + 255) / 256, 256, 0, stream>>>(fc_w1, fcw1_h, 32768);
  cvt_f32_to_f16_kernel<<<(32768 + 255) / 256, 256, 0, stream>>>(fc_w2, fcw2_h, 32768);
  fold_head_kernel<<<1, 64, 0, stream>>>(out_w1, out_b1, out_w2, out_b2, headv, headc);
  dft_tables_kernel<<<16, 256, 0, stream>>>(Cf, Sf, Ainv);

  // --- input projection ---
  input_proj_kernel<<<NPTS / 16, 128, 0, stream>>>(x_in, inw_h, in_b, X);

  // --- layers ---
  for (int l = 0; l < 4; ++l) {
    // spectral conv along y (last axis n): rows=(b,m), transform stride=D
    dft_kernel<<<ROWS2D, 256, 0, stream>>>(X, Fre, Fim, Cf, Sf, N_SZ * D_SZ, D_SZ);
    mode_mix_kernel<<<dim3(ROWS2D / 16, MODES), 128, 0, stream>>>(
        Fre, Fim, Pre, Pim, Wy_re + (size_t)l * 65536, Wy_im + (size_t)l * 65536);
    idft_kernel<<<ROWS2D, 256, 0, stream>>>(Pre, Pim, XS, Ainv, N_SZ * D_SZ, D_SZ, 0);

    // spectral conv along x (axis m): rows=(b,n), transform stride=N*D
    dft_kernel<<<ROWS2D, 256, 0, stream>>>(X, Fre, Fim, Cf, Sf, D_SZ, N_SZ * D_SZ);
    mode_mix_kernel<<<dim3(ROWS2D / 16, MODES), 128, 0, stream>>>(
        Fre, Fim, Pre, Pim, Wx_re + (size_t)l * 65536, Wx_im + (size_t)l * 65536);
    idft_kernel<<<ROWS2D, 256, 0, stream>>>(Pre, Pim, XS, Ainv, D_SZ, N_SZ * D_SZ, 1);

    // fused FF blocks + folded forecast head + x update
    ff_fused_kernel<<<NPTS / (16 * FF_R), 32, 0, stream>>>(
        XS, X, out,
        bcw1_h + (size_t)l * 8192, bc_b1 + (size_t)l * 128,
        bcw2_h + (size_t)l * 8192, bc_b2 + (size_t)l * 64,
        fcw1_h + (size_t)l * 8192, fc_b1 + (size_t)l * 128,
        fcw2_h + (size_t)l * 8192, fc_b2 + (size_t)l * 64,
        headv, headc);
  }
}
